// AdaAug_73598559584738
// MI455X (gfx1250) — matrix operand order repaired and checked by reading, not compiled, not run on hardware
//
#include <hip/hip_runtime.h>

// ---------------------------------------------------------------------------
// Problem constants (match reference)
// ---------------------------------------------------------------------------
#define BS   512
#define NN   500
#define KK   100
#define SS   96
#define HID  32
#define TAU  0.1f

typedef _Float16 v16h __attribute__((ext_vector_type(16)));
typedef _Float16 v8h  __attribute__((ext_vector_type(8)));
typedef float    v8f  __attribute__((ext_vector_type(8)));
typedef float    v4f  __attribute__((ext_vector_type(4)));

// ---------------------------------------------------------------------------
// Kernel 1: bulk copy x -> out (bandwidth bound part, ~196 MB of traffic).
// Native-vector float4 non-temporal streaming copy.
// ---------------------------------------------------------------------------
__global__ __launch_bounds__(256) void adaaug_copy_kernel(
    const float* __restrict__ x, float* __restrict__ out, long n4)
{
    long i = (long)blockIdx.x * blockDim.x + threadIdx.x;
    if (i < n4) {
        const v4f* src = (const v4f*)x;
        v4f*       dst = (v4f*)out;
        v4f v = __builtin_nontemporal_load(&src[i]);
        __builtin_nontemporal_store(v, &dst[i]);
    }
}

// ---------------------------------------------------------------------------
// Kernel 2: gather -> WMMA MLP -> gumbel gate -> scatter rows into out.
// 128 threads = 4 waves; each wave owns a 16-row WMMA tile; block = 64 rows.
// ---------------------------------------------------------------------------
__global__ __launch_bounds__(128) void adaaug_krig_kernel(
    const float* __restrict__ y,
    const float* __restrict__ W1, const float* __restrict__ b1,
    const float* __restrict__ W2, const float* __restrict__ b2,
    const float* __restrict__ mask, const float* __restrict__ gumbel,
    const int*  __restrict__ krig_idx,
    float* __restrict__ out)
{
    __shared__ __align__(32) _Float16 sW1T[HID][SS];  // W1 transposed, f16 (6 KB)
    __shared__ __align__(16) _Float16 sSmpH[64][SS];  // gathered rows, f16 (12 KB)
    __shared__ float  sSmp[64][SS];                   // gathered rows, f32 (24 KB)
    __shared__ int    sSrc[64];                       // source/dest row indices
    __shared__ float  sGb[64];                        // per-row gate value

    const int tid  = threadIdx.x;
    const int base = blockIdx.x * 64;                 // first kriged-row handled here

    // --- source row index per local row (flat dest index == source index) ---
    if (tid < 64) {
        int gr   = base + tid;
        int b    = gr / KK;
        int kk   = gr - b * KK;
        int node = krig_idx[b * KK + kk];
        sSrc[tid] = b * NN + node;
    }
    // --- W1 -> LDS, transposed + f16 ---
    for (int i = tid; i < SS * HID; i += 128) {
        int r = i / HID, c = i - r * HID;             // W1[r][c], row-major (S,HID)
        sW1T[c][r] = (_Float16)W1[i];
    }
    __syncthreads();

    // --- gather smp rows (float4), keep f32 + f16 staging copies ---
    for (int q = tid; q < 64 * (SS / 4); q += 128) {
        int row = q / (SS / 4);
        int c4  = q - row * (SS / 4);
        v4f v = ((const v4f*)(y + (long)sSrc[row] * SS))[c4];
        ((v4f*)&sSmp[row][0])[c4] = v;
        _Float16* hp = &sSmpH[row][c4 * 4];
        hp[0] = (_Float16)v.x; hp[1] = (_Float16)v.y;
        hp[2] = (_Float16)v.z; hp[3] = (_Float16)v.w;
    }
    __syncthreads();

    // --- per-wave 16x32 WMMA tile: h = relu(smp @ W1 + b1) ---
    const int wave    = tid >> 5;
    const int lane    = tid & 31;
    const int half    = lane >> 4;                    // 0/1
    const int n       = lane & 15;                    // A: row M; B/C/D: col N
    const int rowBase = wave * 16;

    v8f acc0 = {};                                    // hidden cols 0..15
    v8f acc1 = {};                                    // hidden cols 16..31
    #pragma unroll
    for (int kt = 0; kt < 3; ++kt) {                  // K = 96 = 3 x 32
        // A fragment: element e <-> K = 32*kt + 16*half + e, row = n
        v8h alo = *(const v8h*)&sSmpH[rowBase + n][32 * kt + 16 * half];
        v8h ahi = *(const v8h*)&sSmpH[rowBase + n][32 * kt + 16 * half + 8];
        v16h a  = __builtin_shufflevector(alo, ahi,
                    0, 1, 2, 3, 4, 5, 6, 7, 8, 9, 10, 11, 12, 13, 14, 15);
        // B fragments: element e <-> K = 32*kt + 16*half + e, col = n (+16)
        v16h bf0 = *(const v16h*)&sW1T[n     ][32 * kt + 16 * half];
        v16h bf1 = *(const v16h*)&sW1T[16 + n][32 * kt + 16 * half];
        acc0 = __builtin_amdgcn_wmma_f32_16x16x32_f16(
                   false, a, false, bf0, (short)0, acc0, false, false);
        acc1 = __builtin_amdgcn_wmma_f32_16x16x32_f16(
                   false, a, false, bf1, (short)0, acc1, false, false);
    }

    // --- bias + relu (C layout: VGPR v, lane-half h -> row M = v + 8h, col n) ---
    float bia0 = b1[n], bia1 = b1[16 + n];
    #pragma unroll
    for (int v = 0; v < 8; ++v) {
        float h0 = acc0[v] + bia0;  acc0[v] = h0 > 0.f ? h0 : 0.f;
        float h1 = acc1[v] + bia1;  acc1[v] = h1 > 0.f ? h1 : 0.f;
    }

    // --- 32 -> 2 layer + gumbel hard-softmax gate (per row) ---
    float w0j0 = W2[2 * n + 0],        w0j1 = W2[2 * n + 1];
    float w1j0 = W2[2 * (16 + n) + 0], w1j1 = W2[2 * (16 + n) + 1];
    float b20 = b2[0], b21 = b2[1];
    #pragma unroll
    for (int v = 0; v < 8; ++v) {
        float p0 = acc0[v] * w0j0 + acc1[v] * w1j0;
        float p1 = acc0[v] * w0j1 + acc1[v] * w1j1;
        #pragma unroll
        for (int m = 1; m < 16; m <<= 1) {            // reduce within 16-lane half
            p0 += __shfl_xor(p0, m, 32);
            p1 += __shfl_xor(p1, m, 32);
        }
        if (n == v) {                                 // one lane per (half, v)
            int   rloc = rowBase + v + 8 * half;
            int   gr   = base + rloc;
            float z0 = (p0 + b20 + gumbel[2 * gr + 0]) * (1.0f / TAU);
            float z1 = (p1 + b21 + gumbel[2 * gr + 1]) * (1.0f / TAU);
            float mz = fmaxf(z0, z1);
            float e0 = __expf(z0 - mz), e1 = __expf(z1 - mz);
            float soft1 = e1 / (e0 + e1);
            float hard1 = (z1 > z0) ? 1.0f : 0.0f;    // tie -> argmax picks idx 0
            sGb[rloc] = (hard1 - soft1) + soft1;      // mirror straight-through fwd
        }
    }
    __syncthreads();

    // --- scatter: out[src][s] = gb1 * mask * smp ---
    for (int q = tid; q < 64 * SS; q += 128) {
        int row = q / SS;
        int col = q - row * SS;
        int gr  = base + row;
        float val = sGb[row] * mask[(long)gr * SS + col] * sSmp[row][col];
        out[(long)sSrc[row] * SS + col] = val;
    }
}

// ---------------------------------------------------------------------------
// Launch: inputs in setup_inputs() dict order:
//   0:x 1:y 2:W1 3:b1 4:W2 5:b2 6:mask 7:gumbel 8:krig_idx 9:idx_of_node
// idx_of_node is the constant N=500 (offsets = b*N), folded into the kernel.
// ---------------------------------------------------------------------------
extern "C" void kernel_launch(void* const* d_in, const int* in_sizes, int n_in,
                              void* d_out, int out_size, void* d_ws, size_t ws_size,
                              hipStream_t stream)
{
    const float* x        = (const float*)d_in[0];
    const float* y        = (const float*)d_in[1];
    const float* W1       = (const float*)d_in[2];
    const float* b1       = (const float*)d_in[3];
    const float* W2       = (const float*)d_in[4];
    const float* b2       = (const float*)d_in[5];
    const float* mask     = (const float*)d_in[6];
    const float* gumbel   = (const float*)d_in[7];
    const int*   krig_idx = (const int*)d_in[8];
    float*       out      = (float*)d_out;

    const long n4 = (long)BS * NN * SS / 4;           // 6,144,000 float4
    adaaug_copy_kernel<<<(int)((n4 + 255) / 256), 256, 0, stream>>>(x, out, n4);

    const int blocks = (BS * KK) / 64;                // 800
    adaaug_krig_kernel<<<blocks, 128, 0, stream>>>(
        y, W1, b1, W2, b2, mask, gumbel, krig_idx, out);
}
